// SingleHeadAttention_2121713844998
// MI455X (gfx1250) — compile-verified, matrix operand order checked by reading, not run on hardware
//
#include <hip/hip_runtime.h>

// ---------------------------------------------------------------------------
// Single-head causal attention, MI455X (gfx1250), bf16 WMMA + fp32 softmax.
//   q = x Wq^T ; k = x Wk^T ; v = x Wv^T   (bf16 WMMA, f32 accum, async-LDS)
//   ctx = softmax(causal(q k^T / sqrt(D))) v  (flash, online softmax, f32)
//   out = ctx Wo^T                            (bf16 WMMA, f32 out)
// ---------------------------------------------------------------------------

typedef __attribute__((ext_vector_type(16))) __bf16 v16bf;
typedef __attribute__((ext_vector_type(8)))  __bf16 v8bf;
typedef __attribute__((ext_vector_type(8)))  float  v8f;

static constexpr int D_MODEL = 512;
static constexpr int SEQ     = 4096;
static constexpr int BATCH   = 4;
static constexpr int MTOT    = BATCH * SEQ;   // 16384 rows

#define DEV __device__ __forceinline__

// Flat pointer -> LDS byte offset. For LDS-aperture flat addresses the low
// 32 bits are the wave-relative LDS address (ISA 10.2),
// which is exactly what GLOBAL_LOAD_ASYNC_TO_LDS wants in its VDST VGPR.
DEV unsigned lds_off(const void* p) { return (unsigned)(unsigned long long)p; }

// ---------------------------------------------------------------------------
// WMMA 16x16x32 bf16 fragment loader (wave32).
// A (16x32): lanes 0-15 hold row m=lane, K in {0..7,16..23}; lanes 16-31 hold
// row m=lane-16, K in {8..15,24..31} (ISA 7.12.2). B mirrored with lane =
// output column; row-major sources with N as rows serve both fragments.
// ---------------------------------------------------------------------------
DEV v16bf load_frag(const __bf16* p, int ld) {
  const int lane = threadIdx.x & 31;
  const int r    = lane & 15;
  const int hk   = (lane >> 4) << 3;            // 0 or 8
  const __bf16* row = p + r * ld + hk;
  v16bf f;
  ((v8bf*)&f)[0] = *(const v8bf*)(row);         // K = hk .. hk+7
  ((v8bf*)&f)[1] = *(const v8bf*)(row + 16);    // K = hk+16 .. hk+23
  return f;
}

// ---------------------------------------------------------------------------
// fp32 -> bf16 elementwise convert
// ---------------------------------------------------------------------------
__global__ void f32_to_bf16(const float* __restrict__ in,
                            __bf16* __restrict__ out, int n) {
  int i = blockIdx.x * 256 + threadIdx.x;
  if (i < n) out[i] = (__bf16)in[i];
}

// ---------------------------------------------------------------------------
// Async stage of one K-slice: X tile 256x32 (4 chunks/thread) and the
// WG-shared W tile 64x32 (1 chunk/thread) -> 5 ASYNCcnt entries per thread.
// ---------------------------------------------------------------------------
DEV void stage_tiles(const __bf16* __restrict__ Xg, const __bf16* __restrict__ Wg,
                     __bf16* sXb, __bf16* sWb, int K, int k0, int tid) {
  const __bf16* xsrc = Xg + k0;
  const __bf16* wsrc = Wg + k0;
#pragma unroll
  for (int c = 0; c < 4; ++c) {
    int idx = tid + (c << 8);                   // 0..1023
    int row = idx >> 2;                         // 0..255
    int kc  = (idx & 3) << 3;                   // 0,8,16,24
    const __bf16* g = xsrc + (long)row * K + kc;
    unsigned l = lds_off(sXb + row * 40 + kc);
    asm volatile("global_load_async_to_lds_b128 %0, %1, off"
                 :: "v"(l), "v"(g) : "memory");
  }
  {
    int row = tid >> 2;                         // 0..63
    int kc  = (tid & 3) << 3;
    const __bf16* g = wsrc + (long)row * K + kc;
    unsigned l = lds_off(sWb + row * 40 + kc);
    asm volatile("global_load_async_to_lds_b128 %0, %1, off"
                 :: "v"(l), "v"(g) : "memory");
  }
}

// ---------------------------------------------------------------------------
// Y[M,N] = X[M,K](bf16) * W[N,K]^T(bf16), f32 accumulate.
// grid = (M/256, N/64), block = 256 (8 waves). Wave w: rows [256*bx+32w,+32)
// x cols [64*by,+64) -> 8 C-frags, 8 WMMA per K-step, 128 WMMAs total.
// Double-buffered async-to-LDS staging of X and W K-slices.
// ---------------------------------------------------------------------------
template <int OUT_BF16>
__global__ __launch_bounds__(256) void
gemm_xwt(const __bf16* __restrict__ X, const __bf16* __restrict__ W,
         void* __restrict__ Yv, int M, int N, int K) {
  constexpr int XS = 40;                         // padded LDS row stride
  __shared__ __attribute__((aligned(16))) __bf16 sX[2][256 * XS];
  __shared__ __attribute__((aligned(16))) __bf16 sW[2][64 * XS];

  const int tid  = threadIdx.x;
  const int wid  = tid >> 5;
  const int lane = tid & 31;
  const int n    = lane & 15;
  const int mb   = (lane >> 4) << 3;
  const int mwg  = blockIdx.x * 256;
  const int n0   = blockIdx.y * 64;
  const int NK   = K >> 5;

  const __bf16* Xg = X + (long)mwg * K;
  const __bf16* Wg = W + (long)n0 * K;

  v8f acc[2][4] = {};

  stage_tiles(Xg, Wg, sX[0], sW[0], K, 0, tid);

  for (int i = 0; i < NK; ++i) {
    const int buf = i & 1;
    __syncthreads();                             // prev buf^1 readers done
    if (i + 1 < NK) {
      stage_tiles(Xg, Wg, sX[buf ^ 1], sW[buf ^ 1], K, (i + 1) << 5, tid);
      asm volatile("s_wait_asynccnt 0x5" ::: "memory");  // buf's 5 retired
    } else {
      asm volatile("s_wait_asynccnt 0x0" ::: "memory");
    }
    __syncthreads();                             // buf visible to all waves

    const __bf16* xb = &sX[buf][0];
    const __bf16* wb = &sW[buf][0];
    v16bf bfr[4];
#pragma unroll
    for (int t = 0; t < 4; ++t) bfr[t] = load_frag(wb + (t * 16) * XS, XS);
#pragma unroll
    for (int h = 0; h < 2; ++h) {
      v16bf a = load_frag(xb + (wid * 32 + h * 16) * XS, XS);
#pragma unroll
      for (int t = 0; t < 4; ++t)
        acc[h][t] = __builtin_amdgcn_wmma_f32_16x16x32_bf16(
            false, a, false, bfr[t], (short)0, acc[h][t], false, false);
    }
  }

#pragma unroll
  for (int h = 0; h < 2; ++h)
#pragma unroll
    for (int t = 0; t < 4; ++t)
#pragma unroll
      for (int j = 0; j < 8; ++j) {
        long idx = (long)(mwg + wid * 32 + h * 16 + mb + j) * N
                 + (n0 + t * 16 + n);
        if (OUT_BF16) ((__bf16*)Yv)[idx] = (__bf16)acc[h][t][j];
        else          ((float*)Yv)[idx]  = acc[h][t][j];
      }
}

// ---------------------------------------------------------------------------
// Flash attention, causal. grid = (SEQ/32, BATCH), block = 256 (8 waves).
//   Phase A: wave (mi,ni) -> 16x16 score tile via WMMA over D=512
//   Phase B: wave 0: rowwise online softmax (m, l, correction), P -> bf16
//   Phase C: wave (mi,ds) -> ctx[16 x 128-slice] += P * V
// V^T staged via register 8x8 transpose (b128 in, b128 out), stride-72 pad.
// ---------------------------------------------------------------------------
__global__ __launch_bounds__(256) void
attn_kernel(const __bf16* __restrict__ Q, const __bf16* __restrict__ K,
            const __bf16* __restrict__ V, __bf16* __restrict__ O) {
  constexpr int QT  = 32;
  constexpr int KT  = 64;
  constexpr int SQS = 66;   // score stride (floats)
  constexpr int SPS = 72;   // P stride (bf16)
  constexpr int SVS = 72;   // V^T stride (bf16)

  __shared__ float  sS[QT * SQS];
  __shared__ __attribute__((aligned(16))) __bf16 sP[QT * SPS];
  __shared__ __attribute__((aligned(16))) __bf16 sVt[D_MODEL * SVS];
  __shared__ float  sM[QT], sL[QT], sCF[QT];

  const int tid  = threadIdx.x;
  const int lane = tid & 31;
  const int wid  = tid >> 5;
  const int q0   = blockIdx.x * QT;
  const long bS  = (long)blockIdx.y * SEQ;

  const int mi  = wid >> 2;        // 16-row half of the query tile
  const int sub = wid & 3;         // phase A: N-tile; phase C: 128-wide slice
  const int n   = lane & 15;
  const int mb  = (lane >> 4) << 3;

  if (tid < QT) { sM[tid] = -3.0e38f; sL[tid] = 0.0f; }

  v8f ctx[8] = {};
  const float rscale = 0.04419417382415922f;   // 1/sqrt(512)
  const int nblk = q0 / KT + 1;

  for (int blk = 0; blk < nblk; ++blk) {
    const int k0 = blk * KT;
    __syncthreads();                           // prev C done with sVt/sP

    // ---- stage V^T: 8x8 register transpose, b128 in / b128 out ----
#pragma unroll
    for (int c = 0; c < 2; ++c) {
      int t    = tid + (c << 8);               // 0..511
      int key8 = (t & 7) << 3;                 // 0..56
      int d0   = (t >> 3) << 3;                // 0..504
      const __bf16* vp = V + (bS + k0 + key8) * D_MODEL + d0;
      v8bf r[8];
#pragma unroll
      for (int j = 0; j < 8; ++j) r[j] = *(const v8bf*)(vp + (long)j * D_MODEL);
#pragma unroll
      for (int i2 = 0; i2 < 8; ++i2) {
        v8bf tr;
#pragma unroll
        for (int j = 0; j < 8; ++j) tr[j] = r[j][i2];
        *(v8bf*)&sVt[(d0 + i2) * SVS + key8] = tr;
      }
    }

    // ---- Phase A: scores ----
    v8f sc = {};
    const __bf16* qptr = Q + (bS + q0 + mi * 16) * D_MODEL;
    const __bf16* kptr = K + (bS + k0 + sub * 16) * D_MODEL;
    for (int kk = 0; kk < D_MODEL; kk += 32) {
      v16bf a = load_frag(qptr + kk, D_MODEL);
      v16bf b = load_frag(kptr + kk, D_MODEL);
      sc = __builtin_amdgcn_wmma_f32_16x16x32_bf16(
          false, a, false, b, (short)0, sc, false, false);
    }
#pragma unroll
    for (int j = 0; j < 8; ++j) {
      int row = mi * 16 + mb + j;
      int qi  = q0 + row;
      int ki  = k0 + sub * 16 + n;
      float val = sc[j] * rscale;
      if (ki > qi) val = -3.0e38f;             // causal mask
      sS[row * SQS + sub * 16 + n] = val;
    }
    __syncthreads();

    // ---- Phase B: online softmax (wave 0, one row per lane) ----
    if (tid < QT) {
      const int row = tid;
      float mOld = sM[row];
      float bm = -3.0e38f;
      for (int c = 0; c < KT; ++c) bm = fmaxf(bm, sS[row * SQS + c]);
      float mNew = fmaxf(mOld, bm);
      float cf   = __expf(mOld - mNew);
      float sum  = 0.0f;
      for (int c = 0; c < KT; ++c) {
        float p = __expf(sS[row * SQS + c] - mNew);
        sP[row * SPS + c] = (__bf16)p;
        sum += p;
      }
      sM[row]  = mNew;
      sCF[row] = cf;
      sL[row]  = sL[row] * cf + sum;
    }
    __syncthreads();

    // ---- Phase C: rescale accumulators, ctx += P * V ----
    float cf[8];
#pragma unroll
    for (int j = 0; j < 8; ++j) cf[j] = sCF[mi * 16 + mb + j];
#pragma unroll
    for (int t = 0; t < 8; ++t)
#pragma unroll
      for (int j = 0; j < 8; ++j) ctx[t][j] *= cf[j];

    for (int kc = 0; kc < KT; kc += 32) {
      v16bf a = load_frag(&sP[(mi * 16) * SPS + kc], SPS);
#pragma unroll
      for (int t = 0; t < 8; ++t) {
        const int d0 = sub * 128 + t * 16;
        v16bf b = load_frag(&sVt[d0 * SVS + kc], SVS);
        ctx[t] = __builtin_amdgcn_wmma_f32_16x16x32_bf16(
            false, a, false, b, (short)0, ctx[t], false, false);
      }
    }
  }

  // ---- finalize: ctx /= l, store bf16 ----
  float linv[8];
#pragma unroll
  for (int j = 0; j < 8; ++j) linv[j] = 1.0f / sL[mi * 16 + mb + j];
#pragma unroll
  for (int t = 0; t < 8; ++t) {
    const int d0 = sub * 128 + t * 16 + n;
#pragma unroll
    for (int j = 0; j < 8; ++j) {
      long row = bS + q0 + mi * 16 + mb + j;
      O[row * D_MODEL + d0] = (__bf16)(ctx[t][j] * linv[j]);
    }
  }
}

// ---------------------------------------------------------------------------
// Launch: converts -> Q/K/V projections -> flash attention -> out projection
// Workspace (bytes): xb@0(16M) qb@16M kb@32M vb@48M ctx@64M
//                    Wq@80M Wk@80.5M Wv@81M Wo@81.5M   (total 82 MiB)
// ---------------------------------------------------------------------------
extern "C" void kernel_launch(void* const* d_in, const int* in_sizes, int n_in,
                              void* d_out, int out_size, void* d_ws,
                              size_t ws_size, hipStream_t stream) {
  (void)in_sizes; (void)n_in; (void)out_size; (void)ws_size;
  const float* x  = (const float*)d_in[0];
  const float* Wq = (const float*)d_in[1];
  const float* Wk = (const float*)d_in[2];
  const float* Wv = (const float*)d_in[3];
  const float* Wo = (const float*)d_in[4];

  char* ws = (char*)d_ws;
  const size_t MB = 1024u * 1024u;
  __bf16* xb   = (__bf16*)(ws + 0 * MB);
  __bf16* qb   = (__bf16*)(ws + 16 * MB);
  __bf16* kb   = (__bf16*)(ws + 32 * MB);
  __bf16* vb   = (__bf16*)(ws + 48 * MB);
  __bf16* ctxb = (__bf16*)(ws + 64 * MB);
  __bf16* wqb  = (__bf16*)(ws + 80 * MB);
  __bf16* wkb  = (__bf16*)(ws + 80 * MB + 512 * 1024);
  __bf16* wvb  = (__bf16*)(ws + 81 * MB);
  __bf16* wob  = (__bf16*)(ws + 81 * MB + 512 * 1024);

  const int nX = MTOT * D_MODEL;
  const int nW = D_MODEL * D_MODEL;
  f32_to_bf16<<<(nX + 255) / 256, 256, 0, stream>>>(x, xb, nX);
  f32_to_bf16<<<(nW + 255) / 256, 256, 0, stream>>>(Wq, wqb, nW);
  f32_to_bf16<<<(nW + 255) / 256, 256, 0, stream>>>(Wk, wkb, nW);
  f32_to_bf16<<<(nW + 255) / 256, 256, 0, stream>>>(Wv, wvb, nW);
  f32_to_bf16<<<(nW + 255) / 256, 256, 0, stream>>>(Wo, wob, nW);

  dim3 gg(MTOT / 256, D_MODEL / 64);
  gemm_xwt<1><<<gg, 256, 0, stream>>>(xb, wqb, qb, MTOT, D_MODEL, D_MODEL);
  gemm_xwt<1><<<gg, 256, 0, stream>>>(xb, wkb, kb, MTOT, D_MODEL, D_MODEL);
  gemm_xwt<1><<<gg, 256, 0, stream>>>(xb, wvb, vb, MTOT, D_MODEL, D_MODEL);

  attn_kernel<<<dim3(SEQ / 32, BATCH), 256, 0, stream>>>(qb, kb, vb, ctxb);

  gemm_xwt<0><<<gg, 256, 0, stream>>>(ctxb, wob, d_out, MTOT, D_MODEL, D_MODEL);
}
